// PaddingMaskedAttention_4587025072516
// MI455X (gfx1250) — matrix operand order repaired and checked by reading, not compiled, not run on hardware
//
#include <hip/hip_runtime.h>

// Problem constants (match reference).
#define BB   2
#define SS   2048
#define HH   1024
#define NHH  16
#define HDIM 64
#define MROWS (BB * SS)   // 4096 total token rows

typedef __attribute__((ext_vector_type(16))) _Float16 v16h;
typedef __attribute__((ext_vector_type(8)))  _Float16 v8h;
typedef __attribute__((ext_vector_type(8)))  float    v8f;

#define NEG_BIG (-3.0e38f)

// Assemble an A fragment (contiguous k = 16*half + s) from two 16B chunks.
static __device__ __forceinline__ v16h make_af(const _Float16* p, int kb) {
  v8h lo = *(const v8h*)(p + 2 * kb);
  v8h hi = *(const v8h*)(p + 2 * kb + 8);
  v16h a;
#pragma unroll
  for (int j = 0; j < 8; ++j) { a[j] = lo[j]; a[8 + j] = hi[j]; }
  return a;
}

// CDNA5 async copy: 16B global -> LDS per lane, tracked by ASYNCcnt.
static __device__ __forceinline__ void async_load_b128(unsigned lds_off,
                                                       const void* gptr) {
  asm volatile("global_load_async_to_lds_b128 %0, %1, off"
               :: "v"(lds_off), "v"((unsigned long long)(uintptr_t)gptr)
               : "memory");
}
static __device__ __forceinline__ void wait_async0() {
  asm volatile("s_wait_asynccnt 0x0" ::: "memory");
}

// ---- kernel 1: input projection  C = X * W^T  (f32 in -> f16 out) -----------
// Block = 8 waves: 128x128 output tile, K-step 32, double-buffered LDS staging.
// Each wave: 16 rows x 128 cols = 8 WMMA accumulators.
// vmode 0: out[b,h,s,d] (Q,K);  vmode 1: out[b,h,d,s] (V transposed for PV).
__global__ __launch_bounds__(256) void proj_kernel(const float* __restrict__ X,
                                                   const float* __restrict__ W,
                                                   _Float16* __restrict__ out,
                                                   int vmode) {
  __shared__ __align__(32) _Float16 Xs[2][128][32];  // 16 KB
  __shared__ __align__(32) _Float16 Ws[2][128][32];  // 16 KB

  const int tid  = threadIdx.x;
  const int lane = tid & 31;
  const int wave = tid >> 5;
  const int n0    = (blockIdx.x & 7) * 128;    // 8 N-strips of 128
  const int mbase = (blockIdx.x >> 3) * 128;   // 32 M-groups of 128
  const int m0    = mbase + wave * 16;
  const int lr   = lane & 15;
  const int kb   = (lane & 16) ? 8 : 0;
  const int koff = (lane & 16) ? 16 : 0;
  const int hi8  = (lane & 16) ? 8 : 0;

  // cooperative staging: each thread moves 16 f32 of X and 16 f32 of W
  const int srow = tid >> 1, shalf = tid & 1;  // 128 rows x 2 halves of 16
  const float* xsrc = X + (size_t)(mbase + srow) * HH + shalf * 16;
  const float* wsrc = W + (size_t)(n0 + srow) * HH + shalf * 16;

  float4 xr[4], wr[4];
  {  // preload K-step 0
    const float4* xp = (const float4*)xsrc;
    xr[0] = xp[0]; xr[1] = xp[1]; xr[2] = xp[2]; xr[3] = xp[3];
    const float4* wp = (const float4*)wsrc;
    wr[0] = wp[0]; wr[1] = wp[1]; wr[2] = wp[2]; wr[3] = wp[3];
  }
  {  // convert + store into buffer 0
    _Float16* xd = &Xs[0][srow][shalf * 16];
    const float* xf = (const float*)xr;
#pragma unroll
    for (int j = 0; j < 16; ++j) xd[j] = (_Float16)xf[j];
    _Float16* wd = &Ws[0][srow][shalf * 16];
    const float* wf = (const float*)wr;
#pragma unroll
    for (int j = 0; j < 16; ++j) wd[j] = (_Float16)wf[j];
  }
  __syncthreads();

  v8f acc[8] = {};
  int buf = 0;
#pragma unroll 1
  for (int k0 = 0; k0 < HH; k0 += 32) {
    const bool more = (k0 + 32) < HH;
    if (more) {  // issue next-step global loads (independent b128s)
      const float4* xp = (const float4*)(xsrc + k0 + 32);
      xr[0] = xp[0]; xr[1] = xp[1]; xr[2] = xp[2]; xr[3] = xp[3];
      const float4* wp = (const float4*)(wsrc + k0 + 32);
      wr[0] = wp[0]; wr[1] = wp[1]; wr[2] = wp[2]; wr[3] = wp[3];
      __builtin_prefetch(xsrc + k0 + 96, 0, 0);
    }

    // compute from LDS[buf]: 1 A-frag feeds 8 WMMAs
    v16h af = make_af(&Xs[buf][wave * 16 + lr][0], kb);
#pragma unroll
    for (int t = 0; t < 8; ++t) {
      v16h bf = *(const v16h*)&Ws[buf][t * 16 + lr][koff];
      acc[t] = __builtin_amdgcn_wmma_f32_16x16x32_f16(
          false, af, false, bf, (short)0, acc[t], false, false);
    }

    if (more) {  // convert + store into the other buffer
      _Float16* xd = &Xs[buf ^ 1][srow][shalf * 16];
      const float* xf = (const float*)xr;
#pragma unroll
      for (int j = 0; j < 16; ++j) xd[j] = (_Float16)xf[j];
      _Float16* wd = &Ws[buf ^ 1][srow][shalf * 16];
      const float* wf = (const float*)wr;
#pragma unroll
      for (int j = 0; j < 16; ++j) wd[j] = (_Float16)wf[j];
    }
    __syncthreads();
    buf ^= 1;
  }

  // C layout: col = lane%16; vgpr i -> row i (+8 for upper lane half).
#pragma unroll
  for (int t = 0; t < 8; ++t) {
    const int n = n0 + t * 16 + lr;
    const int h = n >> 6, d = n & 63;
#pragma unroll
    for (int i = 0; i < 8; ++i) {
      const int m = m0 + i + hi8;
      const int b = m >> 11, s = m & (SS - 1);
      size_t idx = vmode
          ? ((size_t)((b * NHH + h) * HDIM + d) * SS + s)
          : ((size_t)((b * NHH + h) * SS + s) * HDIM + d);
      out[idx] = (_Float16)acc[t][i];
    }
  }
}

// ---- kernel 2: flash attention (online softmax, causal + padding mask) ------
// Block = 8 waves sharing one (b,h); waves own consecutive 16-query tiles.
// K/V chunks staged via CDNA5 async global->LDS copies (ASYNCcnt).
__global__ __launch_bounds__(256) void fattn_kernel(const _Float16* __restrict__ Q,
                                                    const _Float16* __restrict__ K,
                                                    const _Float16* __restrict__ Vt,
                                                    const int* __restrict__ seq_lengths,
                                                    _Float16* __restrict__ O) {
  __shared__ __align__(32) _Float16 Ks[2][32][64];   // 8 KB  (keys x d)
  __shared__ __align__(32) _Float16 Vs[2][64][32];   // 8 KB  (d x keys)
  __shared__ __align__(32) _Float16 Ps[8][16 * 32];  // 8 KB  per-wave P tiles

  const int tid  = threadIdx.x;
  const int lane = tid & 31;
  const int wave = tid >> 5;
  const int bh   = blockIdx.x >> 4;         // 0..31
  const int b    = bh >> 4, h = bh & 15;
  const int qbase = (blockIdx.x & 15) * 128;
  const int q0    = qbase + wave * 16;
  const int seqlen = seq_lengths[b];

  const int lr   = lane & 15;
  const int kb   = (lane & 16) ? 8 : 0;
  const int koff = (lane & 16) ? 16 : 0;
  const int hi8  = (lane & 16) ? 8 : 0;

  const _Float16* Qb = Q  + (size_t)bh * SS * HDIM;
  const _Float16* Kb = K  + (size_t)bh * SS * HDIM;
  const _Float16* Vb = Vt + (size_t)bh * HDIM * SS;
  _Float16* myP = &Ps[wave][0];

  // cooperative staging assignments (one async b128 per thread per tile)
  const int krow = tid >> 3, kseg = tid & 7;   // 32 rows x 8 segs of 8 (d)
  const int vrow = tid >> 2, vseg = tid & 3;   // 64 rows x 4 segs of 8 (keys)
  const _Float16* ksrc = Kb + (size_t)krow * HDIM + kseg * 8;
  const _Float16* vsrc = Vb + (size_t)vrow * SS + vseg * 8;
  // LDS byte offsets (flat shared ptr low 32 bits == LDS offset, ISA 10.2)
  const unsigned ldsK[2] = {(unsigned)(uintptr_t)&Ks[0][krow][kseg * 8],
                            (unsigned)(uintptr_t)&Ks[1][krow][kseg * 8]};
  const unsigned ldsV[2] = {(unsigned)(uintptr_t)&Vs[0][vrow][vseg * 8],
                            (unsigned)(uintptr_t)&Vs[1][vrow][vseg * 8]};

  // Q fragments are loop-invariant: two K-steps over d = 64.
  const _Float16* qrow = Qb + (size_t)(q0 + lr) * HDIM;
  v16h afq0 = make_af(qrow, kb);
  v16h afq1 = make_af(qrow + 32, kb);

  float m[8], l[8];
  v8f acc[4] = {};
#pragma unroll
  for (int i = 0; i < 8; ++i) { m[i] = NEG_BIG; l[i] = 0.f; }

  const int kend_w   = min(q0 + 16, seqlen);          // this wave's key range
  const int kend_blk = min(qbase + 128, seqlen);      // block's key range
  const int nch      = (kend_blk + 31) >> 5;

  // preload chunk 0 (async)
  async_load_b128(ldsK[0], ksrc);
  async_load_b128(ldsV[0], vsrc);
  wait_async0();
  __syncthreads();

  int buf = 0;
#pragma unroll 1
  for (int c = 0; c < nch; ++c) {
    const int kk = c * 32;
    const bool more = (c + 1) < nch;
    if (more) {  // issue next chunk async copies into the other buffer
      async_load_b128(ldsK[buf ^ 1], ksrc + (size_t)(kk + 32) * HDIM);
      async_load_b128(ldsV[buf ^ 1], vsrc + kk + 32);
    }

    if (kk < kend_w) {  // wave-uniform predicate; EXEC stays all-ones inside
      // ---- S = Q K^T for two 16-key tiles (d contraction in 2 steps)
      v8f cc[2] = {};
#pragma unroll
      for (int tt = 0; tt < 2; ++tt) {
        v16h bf0 = *(const v16h*)&Ks[buf][tt * 16 + lr][koff];
        v16h bf1 = *(const v16h*)&Ks[buf][tt * 16 + lr][32 + koff];
        cc[tt] = __builtin_amdgcn_wmma_f32_16x16x32_f16(
            false, afq0, false, bf0, (short)0, cc[tt], false, false);
        cc[tt] = __builtin_amdgcn_wmma_f32_16x16x32_f16(
            false, afq1, false, bf1, (short)0, cc[tt], false, false);
      }

      // ---- scale + causal/padding mask
#pragma unroll
      for (int tt = 0; tt < 2; ++tt) {
        const int key = kk + tt * 16 + lr;
#pragma unroll
        for (int i = 0; i < 8; ++i) {
          const int qr = q0 + i + hi8;
          const float s = cc[tt][i] * 0.125f;  // 1/sqrt(64)
          const bool masked = (key > qr) || (key >= seqlen);
          cc[tt][i] = masked ? NEG_BIG : s;
        }
      }

      // ---- row max over 32 columns (xor-shuffle within lane half)
      float rm[8];
#pragma unroll
      for (int i = 0; i < 8; ++i) {
        float v = fmaxf(cc[0][i], cc[1][i]);
#pragma unroll
        for (int xm = 1; xm <= 8; xm <<= 1)
          v = fmaxf(v, __shfl_xor(v, xm, 32));
        rm[i] = v;
      }

      float alpha[8];
#pragma unroll
      for (int i = 0; i < 8; ++i) {
        const float mn = fmaxf(m[i], rm[i]);
        alpha[i] = __expf(m[i] - mn);
        m[i] = mn;
      }
#pragma unroll
      for (int tt = 0; tt < 2; ++tt)
#pragma unroll
        for (int i = 0; i < 8; ++i)
          cc[tt][i] = __expf(cc[tt][i] - m[i]);

      // ---- row sum + running denominator
#pragma unroll
      for (int i = 0; i < 8; ++i) {
        float v = cc[0][i] + cc[1][i];
#pragma unroll
        for (int xm = 1; xm <= 8; xm <<= 1)
          v += __shfl_xor(v, xm, 32);
        l[i] = l[i] * alpha[i] + v;
      }

      // ---- rescale accumulator
#pragma unroll
      for (int t = 0; t < 4; ++t)
#pragma unroll
        for (int i = 0; i < 8; ++i)
          acc[t][i] *= alpha[i];

      // ---- transpose P (C-layout -> A-layout) via per-wave LDS tile
#pragma unroll
      for (int tt = 0; tt < 2; ++tt)
#pragma unroll
        for (int i = 0; i < 8; ++i)
          myP[(i + hi8) * 32 + tt * 16 + lr] = (_Float16)cc[tt][i];
      asm volatile("s_wait_dscnt 0" ::: "memory");
      v16h afp = make_af(myP + (size_t)lr * 32, kb);

      // ---- O += P V
#pragma unroll
      for (int t = 0; t < 4; ++t) {
        v16h bfv = *(const v16h*)&Vs[buf][t * 16 + lr][koff];
        acc[t] = __builtin_amdgcn_wmma_f32_16x16x32_f16(
            false, afp, false, bfv, (short)0, acc[t], false, false);
      }
    }

    wait_async0();       // own async copies landed in LDS
    __syncthreads();     // visible to all waves
    buf ^= 1;
  }

  // ---- normalize and store O as f16 [B*S][H]
#pragma unroll
  for (int t = 0; t < 4; ++t) {
#pragma unroll
    for (int i = 0; i < 8; ++i) {
      const int r   = q0 + i + hi8;
      const int col = h * HDIM + t * 16 + lr;
      O[(size_t)(b * SS + r) * HH + col] = (_Float16)(acc[t][i] / l[i]);
    }
  }
}

// ---- kernel 3: output projection  out = O * Wo^T  (f16 x f32 -> f32) --------
// Wo staged in double-buffered LDS (128-col strip); A (f16) direct from global.
__global__ __launch_bounds__(256) void oproj_kernel(const _Float16* __restrict__ A,
                                                    const float* __restrict__ W,
                                                    float* __restrict__ out) {
  __shared__ __align__(32) _Float16 Ws[2][128][32];  // 16 KB

  const int tid  = threadIdx.x;
  const int lane = tid & 31;
  const int wave = tid >> 5;
  const int n0 = (blockIdx.x & 7) * 128;
  const int m0 = ((blockIdx.x >> 3) * 8 + wave) * 16;
  const int lr   = lane & 15;
  const int kb   = (lane & 16) ? 8 : 0;
  const int koff = (lane & 16) ? 16 : 0;
  const int hi8  = (lane & 16) ? 8 : 0;

  const int srow = tid >> 1, shalf = tid & 1;
  const float* wsrc = W + (size_t)(n0 + srow) * HH + shalf * 16;
  const _Float16* arow = A + (size_t)(m0 + lr) * HH;

  float4 wr[4];
  {
    const float4* wp = (const float4*)wsrc;
    wr[0] = wp[0]; wr[1] = wp[1]; wr[2] = wp[2]; wr[3] = wp[3];
  }
  {
    _Float16* wd = &Ws[0][srow][shalf * 16];
    const float* wf = (const float*)wr;
#pragma unroll
    for (int j = 0; j < 16; ++j) wd[j] = (_Float16)wf[j];
  }
  __syncthreads();

  v8f acc[8] = {};
  int buf = 0;
#pragma unroll 1
  for (int k0 = 0; k0 < HH; k0 += 32) {
    const bool more = (k0 + 32) < HH;
    if (more) {
      const float4* wp = (const float4*)(wsrc + k0 + 32);
      wr[0] = wp[0]; wr[1] = wp[1]; wr[2] = wp[2]; wr[3] = wp[3];
      __builtin_prefetch(arow + k0 + 96, 0, 0);
    }

    v16h af = make_af(arow + k0, kb);
#pragma unroll
    for (int t = 0; t < 8; ++t) {
      v16h bf = *(const v16h*)&Ws[buf][t * 16 + lr][koff];
      acc[t] = __builtin_amdgcn_wmma_f32_16x16x32_f16(
          false, af, false, bf, (short)0, acc[t], false, false);
    }

    if (more) {
      _Float16* wd = &Ws[buf ^ 1][srow][shalf * 16];
      const float* wf = (const float*)wr;
#pragma unroll
      for (int j = 0; j < 16; ++j) wd[j] = (_Float16)wf[j];
    }
    __syncthreads();
    buf ^= 1;
  }

#pragma unroll
  for (int t = 0; t < 8; ++t) {
    const int n = n0 + t * 16 + lr;
#pragma unroll
    for (int i = 0; i < 8; ++i) {
      const int mrow = m0 + i + hi8;
      out[(size_t)mrow * HH + n] = acc[t][i];
    }
  }
}

// ---- host-side launch -------------------------------------------------------
extern "C" void kernel_launch(void* const* d_in, const int* in_sizes, int n_in,
                              void* d_out, int out_size, void* d_ws, size_t ws_size,
                              hipStream_t stream) {
  (void)in_sizes; (void)n_in; (void)out_size; (void)ws_size;
  const float* x   = (const float*)d_in[0];
  const int*   seq = (const int*)d_in[1];
  const float* Wq  = (const float*)d_in[2];
  const float* Wk  = (const float*)d_in[3];
  const float* Wv  = (const float*)d_in[4];
  const float* Wo  = (const float*)d_in[5];
  float* out = (float*)d_out;

  // Workspace: Q, K (f16 [b,h,s,d]), Vt (f16 [b,h,d,s]), O (f16 [B*S,H]) = 32 MB.
  _Float16* ws = (_Float16*)d_ws;
  const size_t SZ = (size_t)MROWS * HH;
  _Float16* Qw = ws;
  _Float16* Kw = ws + SZ;
  _Float16* Vw = ws + 2 * SZ;
  _Float16* Ow = ws + 3 * SZ;

  dim3 blk(256);
  proj_kernel<<<256, blk, 0, stream>>>(x, Wq, Qw, 0);
  proj_kernel<<<256, blk, 0, stream>>>(x, Wk, Kw, 0);
  proj_kernel<<<256, blk, 0, stream>>>(x, Wv, Vw, 1);
  fattn_kernel<<<512, blk, 0, stream>>>(Qw, Kw, Vw, seq, Ow);
  oproj_kernel<<<256, blk, 0, stream>>>(Ow, Wo, out);
}
